// BASE_RelPatchAttention3D_TCHW_21732534517872
// MI455X (gfx1250) — matrix operand order repaired and checked by reading, not compiled, not run on hardware
//
#include <hip/hip_runtime.h>

#define B_ 2
#define T_ 8
#define C_ 128
#define H_ 128
#define W_ 128
#define HW_ (H_*W_)
#define CHW_ (C_*H_*W_)
#define NTOK 512          // T * 4*4*4 patch tokens per batch
#define PC 32768          // 32*32*32 patch content
#define SPLITK 8
#define KCHUNK (PC / SPLITK)   // 4096
#define KSTEP 32
#define TILE 64
#define NTILE (NTOK / TILE)    // 8
#define SMOOTH 1e-5f

typedef __attribute__((ext_vector_type(16))) __bf16 bf16x16;
typedef __attribute__((ext_vector_type(8)))  float  f32x8;
typedef __attribute__((ext_vector_type(4)))  unsigned int u32x4;
typedef __attribute__((ext_vector_type(8)))  int  i32x8;
typedef __attribute__((ext_vector_type(4)))  int  i32x4;

union Pack16 { __bf16 h[16]; bf16x16 v; };

// patchified row base address in the original [B,T,C,H,W] tensor
__device__ __forceinline__ long row_base(int b, int tok) {
  int t  = tok >> 6;
  int s0 = (tok >> 4) & 3;
  int s1 = (tok >> 2) & 3;
  int s2 = tok & 3;
  return (long)(b * T_ + t) * CHW_ + (long)(s0 * 32) * HW_ + (s1 * 32) * W_ + s2 * 32;
}

// Issue one TDM load of a 64-token x 32-float tile:
// 4D tile [ww=32, s2=4, s1=4, s0=4], strides (elements): 32, 4096(=32*W), 524288(=32*H*W)
// -> lands contiguous [64][32] f32 in LDS (iteration order s0,s1,s2,ww == tile-row order).
__device__ __forceinline__ void tdm_load_tile(unsigned int lds_addr, const float* gptr) {
  unsigned long long ga = (unsigned long long)(uintptr_t)gptr;
  u32x4 g0;
  g0[0] = 1u;                                          // count=1, user descriptor
  g0[1] = lds_addr;                                    // lds_addr (bytes)
  g0[2] = (unsigned int)ga;                            // global_addr[31:0]
  g0[3] = (unsigned int)((ga >> 32) & 0x1FFFFFFu) | (2u << 30);  // addr[56:32] | type=2

  i32x8 g1;
  g1[0] = (int)(2u << 16);        // workgroup_mask=0, data_size=2 (4 bytes)
  g1[1] = (int)(32u << 16);       // tensor_dim0 = 32 (bits 79:48, low half)
  g1[2] = (int)(4u << 16);        // tensor_dim0 hi=0 | tensor_dim1 = 4 (bits 111:80, low half)
  g1[3] = (int)(32u << 16);       // tensor_dim1 hi=0 | tile_dim0 = 32 (bits 127:112)
  g1[4] = (int)(4u | (4u << 16)); // tile_dim1 = 4, tile_dim2 = 4
  g1[5] = (int)32u;               // tensor_dim0_stride = 32 (low 32 of 48)
  g1[6] = (int)(4096u << 16);     // dim0_stride hi=0 | tensor_dim1_stride lo16 = 4096
  g1[7] = 0;                      // tensor_dim1_stride hi = 0

  i32x4 g2;
  g2[0] = 4;                      // tensor_dim2 = 4
  g2[1] = 4;                      // tensor_dim3 = 4
  g2[2] = (int)524288u;           // tensor_dim2_stride = 32*H*W (low 32 of 48)
  g2[3] = (int)(4u << 16);        // dim2_stride hi=0 | tile_dim3 = 4

  i32x4 g3;
  g3[0] = 0; g3[1] = 0; g3[2] = 0; g3[3] = 0;  // dim3 stride / dim4 unused

  i32x8 g4;                       // extra operand of the 6-arg builtin form: zero-filled
  g4[0] = 0; g4[1] = 0; g4[2] = 0; g4[3] = 0;
  g4[4] = 0; g4[5] = 0; g4[6] = 0; g4[7] = 0;

  __builtin_amdgcn_tensor_load_to_lds(g0, g1, g2, g3, g4, 0);
}

// gather 16 f32 from an LDS tile row at element offsets [o0..o0+7] and [o1..o1+7],
// convert to a bf16x16 WMMA fragment (v_cvt_pk_bf16_f32)
__device__ __forceinline__ bf16x16 cvt_frag(const float* row, int o0, int o1) {
  float4 a0 = *(const float4*)(row + o0);
  float4 a1 = *(const float4*)(row + o0 + 4);
  float4 a2 = *(const float4*)(row + o1);
  float4 a3 = *(const float4*)(row + o1 + 4);
  Pack16 p;
  p.h[0]  = (__bf16)a0.x; p.h[1]  = (__bf16)a0.y; p.h[2]  = (__bf16)a0.z; p.h[3]  = (__bf16)a0.w;
  p.h[4]  = (__bf16)a1.x; p.h[5]  = (__bf16)a1.y; p.h[6]  = (__bf16)a1.z; p.h[7]  = (__bf16)a1.w;
  p.h[8]  = (__bf16)a2.x; p.h[9]  = (__bf16)a2.y; p.h[10] = (__bf16)a2.z; p.h[11] = (__bf16)a2.w;
  p.h[12] = (__bf16)a3.x; p.h[13] = (__bf16)a3.y; p.h[14] = (__bf16)a3.z; p.h[15] = (__bf16)a3.w;
  return p.v;
}

// ---------------- row sum-of-squares (qq / kk) ----------------
__global__ __launch_bounds__(256) void norms_kernel(const float* __restrict__ q,
                                                    const float* __restrict__ k,
                                                    float* __restrict__ nq,
                                                    float* __restrict__ nk) {
  __shared__ float red[256];
  int blk   = blockIdx.x;          // [0, 2048)
  int which = blk >> 10;           // 0 = q, 1 = k
  int r     = blk & 1023;
  int b     = r >> 9;
  int tok   = r & 511;
  const float* src = which ? k : q;
  float*       dst = which ? nk : nq;
  long base = row_base(b, tok);
  int tid = threadIdx.x;
  float s = 0.f;
  for (int g = tid; g < 1024; g += 256) {      // g = cc*32 + hh, 32 contiguous floats each
    long gb = base + (long)(g >> 5) * HW_ + (g & 31) * W_;
    #pragma unroll
    for (int j = 0; j < 8; ++j) {
      float4 v = *(const float4*)(src + gb + j * 4);
      s += v.x * v.x + v.y * v.y + v.z * v.z + v.w * v.w;
    }
  }
  red[tid] = s;
  __syncthreads();
  for (int o = 128; o > 0; o >>= 1) {
    if (tid < o) red[tid] += red[tid + o];
    __syncthreads();
  }
  if (tid == 0) dst[b * NTOK + tok] = red[0];
}

// -------- Qf·Kf^T partial GEMM: TDM-staged, double-buffered, bf16 WMMA, split-K --------
__global__ __launch_bounds__(256) void gemm_qk_kernel(const float* __restrict__ q,
                                                      const float* __restrict__ k,
                                                      float* __restrict__ part) {
  __shared__ __align__(16) float qlds[2][TILE * KSTEP];   // 2 x 8 KB f32 tiles
  __shared__ __align__(16) float klds[2][TILE * KSTEP];

  int blk  = blockIdx.x;
  int sk   = blk & (SPLITK - 1);
  int tile = blk / SPLITK;
  int tn   = tile & (NTILE - 1);
  int tm   = (tile / NTILE) & (NTILE - 1);
  int b    = tile / (NTILE * NTILE);

  int tid    = threadIdx.x;
  int lane   = tid & 31;
  int wave   = tid >> 5;       // 8 waves
  int mw     = wave >> 1;      // 4 M-subtiles of 16
  int nh     = wave & 1;       // 2 N-halves of 32
  int lane16 = lane & 15;
  int hi     = lane >> 4;

  // tile row 0 has s0=s1=s2=0 -> base = (b*T + t) * CHW
  long bq = (long)(b * T_ + tm) * CHW_;
  long bk = (long)(b * T_ + tn) * CHW_;

  unsigned int q_lds[2], k_lds[2];
  q_lds[0] = (unsigned int)(uintptr_t)&qlds[0][0];
  q_lds[1] = (unsigned int)(uintptr_t)&qlds[1][0];
  k_lds[0] = (unsigned int)(uintptr_t)&klds[0][0];
  k_lds[1] = (unsigned int)(uintptr_t)&klds[1][0];

  f32x8 c0 = {};
  f32x8 c1 = {};

  int k0base = sk * KCHUNK;
  const int NSTEP = KCHUNK / KSTEP;   // 128

  // prologue: stage step 0 into buffer 0
  if (wave == 0) {
    int cc = k0base >> 10, hh = (k0base >> 5) & 31;
    long off = (long)cc * HW_ + hh * W_;
    tdm_load_tile(q_lds[0], q + bq + off);
    tdm_load_tile(k_lds[0], k + bk + off);
  }

  for (int step = 0; step < NSTEP; ++step) {
    int cur = step & 1;
    if (wave == 0) __builtin_amdgcn_s_wait_tensorcnt(0);  // current buffer's DMA done
    __syncthreads();                                      // all waves: LDS filled, prev reads done
    if (wave == 0 && step + 1 < NSTEP) {                  // overlap next DMA with compute
      int kn = k0base + (step + 1) * KSTEP;
      int cc = kn >> 10, hh = (kn >> 5) & 31;
      long off = (long)cc * HW_ + hh * W_;
      tdm_load_tile(q_lds[cur ^ 1], q + bq + off);
      tdm_load_tile(k_lds[cur ^ 1], k + bk + off);
    }

    const float* qrow  = &qlds[cur][(mw * 16 + lane16) * KSTEP];
    const float* krow0 = &klds[cur][(nh * 32 + lane16) * KSTEP];
    const float* krow1 = krow0 + 16 * KSTEP;

    // A: K split {0..7,16..23} / {8..15,24..31};  B: K split {0..7,16..23} / {8..15,24..31}
    bf16x16 a  = cvt_frag(qrow,  hi * 8,  hi * 8 + 16);
    bf16x16 b0 = cvt_frag(krow0, hi * 8,  hi * 8 + 16);
    bf16x16 b1 = cvt_frag(krow1, hi * 8,  hi * 8 + 16);

    c0 = __builtin_amdgcn_wmma_f32_16x16x32_bf16(false, a, false, b0, (short)0, c0, false, false);
    c1 = __builtin_amdgcn_wmma_f32_16x16x32_bf16(false, a, false, b1, (short)0, c1, false, false);
  }

  // store split-K partial tile: part[sk][b][M][N]
  long pbase = (((long)sk * B_ + b) * NTOK) * NTOK;
  int Mb = tm * TILE + mw * 16 + hi * 8;
  int N0 = tn * TILE + nh * 32 + lane16;
  #pragma unroll
  for (int r2 = 0; r2 < 8; ++r2) {
    part[pbase + (long)(Mb + r2) * NTOK + N0]      = c0[r2];
    part[pbase + (long)(Mb + r2) * NTOK + N0 + 16] = c1[r2];
  }
}

// ---------------- split-K reduce + Tanimoto nonlinearity + column sum ----------------
__global__ __launch_bounds__(256) void attn_kernel(const float* __restrict__ part,
                                                   const float* __restrict__ nq,
                                                   const float* __restrict__ nk,
                                                   float* __restrict__ attn) {
  int gid = blockIdx.x * 256 + threadIdx.x;  // [0, B*512)
  int b = gid >> 9;
  int m = gid & 511;
  const float inv = 1.0f / (float)PC;        // scale^2
  float kkv = nk[b * NTOK + m] * inv;
  float acc = 0.f;
  for (int n = 0; n < NTOK; ++n) {
    float s = 0.f;
    #pragma unroll
    for (int sk = 0; sk < SPLITK; ++sk)
      s += part[(((long)sk * B_ + b) * NTOK + n) * NTOK + m];
    float qk  = s * inv;
    float qq  = nq[b * NTOK + n] * inv;
    float den = qq + kkv - qk + SMOOTH;
    acc += __expf(__logf(qk + SMOOTH) - __logf(den));
  }
  attn[b * NTOK + m] = acc;
}

// ---------------- out = attn[token(idx)] * v (unpatchify collapses to elementwise) ----
__global__ __launch_bounds__(256) void scale_v_kernel(const float* __restrict__ v,
                                                      const float* __restrict__ attn,
                                                      float* __restrict__ out) {
  long gid = (long)blockIdx.x * 256 + threadIdx.x;
  long idx = gid * 4;
  int ww = (int)(idx & 127);
  int hh = (int)((idx >> 7) & 127);
  int cc = (int)((idx >> 14) & 127);
  int t  = (int)((idx >> 21) & 7);
  int b  = (int)(idx >> 24);
  int tok = (t << 6) | ((cc >> 5) << 4) | ((hh >> 5) << 2) | (ww >> 5);
  float a = attn[b * NTOK + tok];
  float4 vv = *(const float4*)(v + idx);
  float4 o = {a * vv.x, a * vv.y, a * vv.z, a * vv.w};
  *(float4*)(out + idx) = o;
}

extern "C" void kernel_launch(void* const* d_in, const int* in_sizes, int n_in,
                              void* d_out, int out_size, void* d_ws, size_t ws_size,
                              hipStream_t stream) {
  const float* q = (const float*)d_in[0];
  const float* k = (const float*)d_in[1];
  const float* v = (const float*)d_in[2];
  float* out = (float*)d_out;

  float* part = (float*)d_ws;                              // SPLITK*B*512*512 f32 = 16 MB
  size_t part_n = (size_t)SPLITK * B_ * NTOK * NTOK;
  float* nq = part + part_n;                               // B*512
  float* nk = nq + B_ * NTOK;                              // B*512
  float* at = nk + B_ * NTOK;                              // B*512

  norms_kernel<<<2 * B_ * NTOK, 256, 0, stream>>>(q, k, nq, nk);               // 2048 blocks
  gemm_qk_kernel<<<B_ * NTILE * NTILE * SPLITK, 256, 0, stream>>>(q, k, part); // 1024 blocks
  attn_kernel<<<(B_ * NTOK) / 256, 256, 0, stream>>>(part, nq, nk, at);        // 4 blocks
  scale_v_kernel<<<(B_ * (long)T_ * C_ * H_ * W_ / 4) / 256, 256, 0, stream>>>(v, at, out);
}
